// MultiHeadAttention_5875515261428
// MI455X (gfx1250) — compile-verified, matrix operand order checked by reading, not run on hardware
//
#include <hip/hip_runtime.h>

// ---------------------------------------------------------------------------
// MHA for B=2, T=2048, C=1024, H=16, D=64 on gfx1250 (wave32, WMMA f16).
// Fragments = two contiguous 16B LDS chunks; global->LDS staging uses the
// CDNA5 async path (global_load_async_to_lds_b128 / s_wait_asynccnt).
// ---------------------------------------------------------------------------

typedef __attribute__((ext_vector_type(16))) _Float16 v16h;
typedef __attribute__((ext_vector_type(8)))  _Float16 v8h;
typedef __attribute__((ext_vector_type(8)))  float    v8f;

constexpr int Bc = 2, Tc = 2048, Cc = 1024, Hc = 16, Dc = 64;
constexpr int Mtot = Bc * Tc;      // 4096
constexpr int N3C  = 3 * Cc;       // 3072

// ---------------- WMMA fragment helpers (16x16x32 f16, wave32) -------------
__device__ __forceinline__ v16h load_frag_rowk(const _Float16* p, int ld,
                                               int major_base, int k_base,
                                               int lane) {
  const int mj = major_base + (lane & 15);
  const int h8 = ((lane >> 4) & 1) * 8;
  const _Float16* q = p + (size_t)mj * ld + k_base + h8;
  v8h lo = *(const v8h*)q;
  v8h hi = *(const v8h*)(q + 16);
  return __builtin_shufflevector(lo, hi, 0, 1, 2, 3, 4, 5, 6, 7,
                                 8, 9, 10, 11, 12, 13, 14, 15);
}

__device__ __forceinline__ v8f wmma16(v16h a, v16h b, v8f c) {
  return __builtin_amdgcn_wmma_f32_16x16x32_f16(false, a, false, b,
                                                (short)0, c, false, false);
}

// CDNA5 async global->LDS copy of one 16B chunk per lane (ASYNCcnt tracked).
__device__ __forceinline__ void async_copy_b128(void* lds_dst,
                                                const void* gsrc) {
  unsigned dst = (unsigned)(uintptr_t)lds_dst;   // low 32 bits = LDS offset
  asm volatile("global_load_async_to_lds_b128 %0, %1, off"
               :: "v"(dst), "v"(gsrc) : "memory");
}
__device__ __forceinline__ void wait_async0() {
  asm volatile("s_wait_asynccnt 0" ::: "memory");
}

__device__ __forceinline__ float redmax16(float x) {
#pragma unroll
  for (int off = 1; off < 16; off <<= 1) x = fmaxf(x, __shfl_xor(x, off, 32));
  return x;
}
__device__ __forceinline__ float redsum16(float x) {
#pragma unroll
  for (int off = 1; off < 16; off <<= 1) x += __shfl_xor(x, off, 32);
  return x;
}

// ---------------------------- fp32 -> f16 conversion -----------------------
__global__ void cvt_f32_f16(const float* __restrict__ in,
                            _Float16* __restrict__ out, int n) {
  for (int i = blockIdx.x * blockDim.x + threadIdx.x; i < n;
       i += gridDim.x * blockDim.x)
    out[i] = (_Float16)in[i];
}

// ----------------------------- QKV projection GEMM -------------------------
__global__ __launch_bounds__(256) void qkv_gemm(
    const _Float16* __restrict__ X, const _Float16* __restrict__ W,
    const float* __restrict__ bias, _Float16* __restrict__ Qh,
    _Float16* __restrict__ Kh, _Float16* __restrict__ Vh) {
  __shared__ _Float16 As[128][40];    // A tile:  128 rows x 32 k (+8 pad)
  __shared__ _Float16 BsT[128][40];   // B tile transposed: 128 n x 32 k (+8)

  const int tid  = threadIdx.x;
  const int lane = tid & 31;
  const int wave = tid >> 5;        // 0..7
  const int wm   = wave & 3;        // 4 waves along M (32 rows each)
  const int wn   = wave >> 2;       // 2 waves along N (64 cols each)
  const int m0   = blockIdx.x * 128;
  const int n0   = blockIdx.y * 128;

  const int ra = tid >> 2, ca = (tid & 3) * 8;   // A chunk (this + +64 row)
  const int rb = tid >> 4, cb = (tid & 15) * 8;  // B chunk (this + +16 row)

  v8f acc[2][4];
#pragma unroll
  for (int a = 0; a < 2; ++a)
#pragma unroll
    for (int b = 0; b < 4; ++b) acc[a][b] = (v8f)0.0f;

  for (int kk = 0; kk < Cc; kk += 32) {
    // A tile: async global->LDS, 2 chunks per thread
    async_copy_b128(&As[ra][ca], &X[(size_t)(m0 + ra) * Cc + kk + ca]);
    async_copy_b128(&As[ra + 64][ca], &X[(size_t)(m0 + ra + 64) * Cc + kk + ca]);
    // B tile: both global loads in flight, then transpose-scatter to LDS
    v8h w0 = *(const v8h*)&W[(size_t)(kk + rb) * N3C + n0 + cb];
    v8h w1 = *(const v8h*)&W[(size_t)(kk + rb + 16) * N3C + n0 + cb];
#pragma unroll
    for (int j = 0; j < 8; ++j) BsT[cb + j][rb] = w0[j];
#pragma unroll
    for (int j = 0; j < 8; ++j) BsT[cb + j][rb + 16] = w1[j];
    if (kk + 32 < Cc)
      __builtin_prefetch(&W[(size_t)(kk + 32 + rb) * N3C + n0 + cb], 0, 1);
    wait_async0();
    __syncthreads();

    v16h af[2], bf[4];
#pragma unroll
    for (int mi = 0; mi < 2; ++mi)
      af[mi] = load_frag_rowk(&As[0][0], 40, wm * 32 + mi * 16, 0, lane);
#pragma unroll
    for (int ni = 0; ni < 4; ++ni)
      bf[ni] = load_frag_rowk(&BsT[0][0], 40, wn * 64 + ni * 16, 0, lane);
#pragma unroll
    for (int mi = 0; mi < 2; ++mi)
#pragma unroll
      for (int ni = 0; ni < 4; ++ni)
        acc[mi][ni] = wmma16(af[mi], bf[ni], acc[mi][ni]);
    __syncthreads();
  }

  const int h  = (lane >> 4) & 1;
  const int ln = lane & 15;
#pragma unroll
  for (int mi = 0; mi < 2; ++mi)
#pragma unroll
    for (int ni = 0; ni < 4; ++ni)
#pragma unroll
      for (int i = 0; i < 8; ++i) {
        int m = m0 + wm * 32 + mi * 16 + i + 8 * h;  // 0..4095
        int n = n0 + wn * 64 + ni * 16 + ln;         // 0..3071
        float val = acc[mi][ni][i] + bias[n];
        int region = n >> 10;           // 0=q 1=k 2=v
        int cc = n & 1023;
        int head = cc >> 6, d = cc & 63;
        int bb = m >> 11, t = m & 2047;
        size_t idx = ((size_t)(bb * Hc + head) * Tc + t) * Dc + d;
        if (region == 0)      Qh[idx] = (_Float16)(val * 0.125f);  // 1/sqrt(64)
        else if (region == 1) Kh[idx] = (_Float16)val;
        else                  Vh[idx] = (_Float16)val;
      }
}

// ------------------------------ Flash attention ----------------------------
// grid: (B*H, T/128); block: 256 threads = 8 waves; wave owns 16 query rows.
__global__ __launch_bounds__(256) void attn_fwd(
    const _Float16* __restrict__ Qh, const _Float16* __restrict__ Kh,
    const _Float16* __restrict__ Vh, _Float16* __restrict__ Oh) {
  __shared__ _Float16 Kt[64][72];        // [key][d]   (B operand of Q@K^T)
  __shared__ _Float16 VtT[64][72];       // [d][key]   (B operand of P@V)
  __shared__ _Float16 Ps[8][16][72];     // per-wave P tile [qrow][key]

  const int bh   = blockIdx.x;   // 0..31
  const int qb   = blockIdx.y;   // 0..15
  const int tid  = threadIdx.x;
  const int lane = tid & 31;
  const int wave = tid >> 5;     // 0..7
  const int h    = (lane >> 4) & 1;
  const int ln   = lane & 15;

  const _Float16* Qb = Qh + (size_t)bh * Tc * Dc;
  const _Float16* Kb = Kh + (size_t)bh * Tc * Dc;
  const _Float16* Vb = Vh + (size_t)bh * Tc * Dc;

  const int qrow0 = qb * 128 + wave * 16;   // wave's query base (in T)

  v16h qf[2];
#pragma unroll
  for (int ks = 0; ks < 2; ++ks)
    qf[ks] = load_frag_rowk(Qb, Dc, qrow0, ks * 32, lane);

  v8f accd[4];
#pragma unroll
  for (int j = 0; j < 4; ++j) accd[j] = (v8f)0.0f;
  float mrow[8], lrow[8];
#pragma unroll
  for (int i = 0; i < 8; ++i) { mrow[i] = -1e30f; lrow[i] = 0.0f; }

  const int rr = tid >> 3, cc = (tid & 7) * 8;  // chunk (this + +32 row)

  const int ntiles = 2 * qb + 2;           // causal: only lower-tri k-tiles
  for (int kt = 0; kt < ntiles; ++kt) {
    const int kbase = kt * 64;
    const bool active = (kbase <= qrow0 + 15);  // wave-uniform

    // K tile: async global->LDS (no transpose needed)
    async_copy_b128(&Kt[rr][cc], &Kb[(size_t)(kbase + rr) * Dc + cc]);
    async_copy_b128(&Kt[rr + 32][cc], &Kb[(size_t)(kbase + rr + 32) * Dc + cc]);
    // V tile: both loads in flight, then transpose-scatter
    v8h v0 = *(const v8h*)&Vb[(size_t)(kbase + rr) * Dc + cc];
    v8h v1 = *(const v8h*)&Vb[(size_t)(kbase + rr + 32) * Dc + cc];
#pragma unroll
    for (int j = 0; j < 8; ++j) VtT[cc + j][rr] = v0[j];
#pragma unroll
    for (int j = 0; j < 8; ++j) VtT[cc + j][rr + 32] = v1[j];
    wait_async0();
    __syncthreads();

    if (active) {
      // S = Q @ K^T  (16 x 64 per wave); B[k=d][n=key] from Kt[key][d]
      v8f sf[4];
#pragma unroll
      for (int j = 0; j < 4; ++j) sf[j] = (v8f)0.0f;
#pragma unroll
      for (int j = 0; j < 4; ++j)
#pragma unroll
        for (int ks = 0; ks < 2; ++ks) {
          v16h bf = load_frag_rowk(&Kt[0][0], 72, j * 16, ks * 32, lane);
          sf[j] = wmma16(qf[ks], bf, sf[j]);
        }

      if (kbase + 63 > qrow0) {            // tile touches the diagonal
#pragma unroll
        for (int j = 0; j < 4; ++j)
#pragma unroll
          for (int i = 0; i < 8; ++i) {
            int col = kbase + j * 16 + ln;
            int row = qrow0 + i + 8 * h;
            if (col > row) sf[j][i] = -1e30f;
          }
      }

      // online softmax (rows live in 16-lane halves)
#pragma unroll
      for (int i = 0; i < 8; ++i) {
        float s = sf[0][i];
#pragma unroll
        for (int j = 1; j < 4; ++j) s = fmaxf(s, sf[j][i]);
        s = redmax16(s);
        float mn = fmaxf(mrow[i], s);
        float alpha = __expf(mrow[i] - mn);
        mrow[i] = mn;
        float rs = 0.0f;
#pragma unroll
        for (int j = 0; j < 4; ++j) {
          float p = __expf(sf[j][i] - mn);
          sf[j][i] = p;
          rs += p;
        }
        rs = redsum16(rs);
        lrow[i] = lrow[i] * alpha + rs;
#pragma unroll
        for (int j = 0; j < 4; ++j) accd[j][i] = accd[j][i] * alpha;
      }

      // P -> LDS (C-layout -> A-layout), per-wave buffer
#pragma unroll
      for (int j = 0; j < 4; ++j)
#pragma unroll
        for (int i = 0; i < 8; ++i)
          Ps[wave][i + 8 * h][j * 16 + ln] = (_Float16)sf[j][i];
    }
    __syncthreads();

    if (active) {
      // O += P @ V;  A = Ps[qrow][key], B[k=key][n=d] from VtT[d][key]
#pragma unroll
      for (int ks = 0; ks < 2; ++ks) {
        v16h pf = load_frag_rowk(&Ps[wave][0][0], 72, 0, ks * 32, lane);
#pragma unroll
        for (int j = 0; j < 4; ++j) {
          v16h vf = load_frag_rowk(&VtT[0][0], 72, j * 16, ks * 32, lane);
          accd[j] = wmma16(pf, vf, accd[j]);
        }
      }
    }
    __syncthreads();
  }

  const int bb = bh / Hc, hd = bh % Hc;
#pragma unroll
  for (int j = 0; j < 4; ++j)
#pragma unroll
    for (int i = 0; i < 8; ++i) {
      int t = qrow0 + i + 8 * h;
      float val = accd[j][i] / lrow[i];
      Oh[((size_t)(bb * Tc + t)) * Cc + hd * Dc + j * 16 + ln] = (_Float16)val;
    }
}

// ------------------------------ Output projection --------------------------
__global__ __launch_bounds__(256) void proj_gemm(
    const _Float16* __restrict__ X, const _Float16* __restrict__ W,
    const float* __restrict__ bias, float* __restrict__ out) {
  __shared__ _Float16 As[128][40];
  __shared__ _Float16 BsT[128][40];

  const int tid  = threadIdx.x;
  const int lane = tid & 31;
  const int wave = tid >> 5;
  const int wm   = wave & 3;
  const int wn   = wave >> 2;
  const int m0   = blockIdx.x * 128;
  const int n0   = blockIdx.y * 128;

  const int ra = tid >> 2, ca = (tid & 3) * 8;
  const int rb = tid >> 4, cb = (tid & 15) * 8;

  v8f acc[2][4];
#pragma unroll
  for (int a = 0; a < 2; ++a)
#pragma unroll
    for (int b = 0; b < 4; ++b) acc[a][b] = (v8f)0.0f;

  for (int kk = 0; kk < Cc; kk += 32) {
    async_copy_b128(&As[ra][ca], &X[(size_t)(m0 + ra) * Cc + kk + ca]);
    async_copy_b128(&As[ra + 64][ca], &X[(size_t)(m0 + ra + 64) * Cc + kk + ca]);
    v8h w0 = *(const v8h*)&W[(size_t)(kk + rb) * Cc + n0 + cb];
    v8h w1 = *(const v8h*)&W[(size_t)(kk + rb + 16) * Cc + n0 + cb];
#pragma unroll
    for (int j = 0; j < 8; ++j) BsT[cb + j][rb] = w0[j];
#pragma unroll
    for (int j = 0; j < 8; ++j) BsT[cb + j][rb + 16] = w1[j];
    if (kk + 32 < Cc)
      __builtin_prefetch(&W[(size_t)(kk + 32 + rb) * Cc + n0 + cb], 0, 1);
    wait_async0();
    __syncthreads();

    v16h af[2], bf[4];
#pragma unroll
    for (int mi = 0; mi < 2; ++mi)
      af[mi] = load_frag_rowk(&As[0][0], 40, wm * 32 + mi * 16, 0, lane);
#pragma unroll
    for (int ni = 0; ni < 4; ++ni)
      bf[ni] = load_frag_rowk(&BsT[0][0], 40, wn * 64 + ni * 16, 0, lane);
#pragma unroll
    for (int mi = 0; mi < 2; ++mi)
#pragma unroll
      for (int ni = 0; ni < 4; ++ni)
        acc[mi][ni] = wmma16(af[mi], bf[ni], acc[mi][ni]);
    __syncthreads();
  }

  const int h  = (lane >> 4) & 1;
  const int ln = lane & 15;
#pragma unroll
  for (int mi = 0; mi < 2; ++mi)
#pragma unroll
    for (int ni = 0; ni < 4; ++ni)
#pragma unroll
      for (int i = 0; i < 8; ++i) {
        int m = m0 + wm * 32 + mi * 16 + i + 8 * h;
        int n = n0 + wn * 64 + ni * 16 + ln;
        out[(size_t)m * Cc + n] = acc[mi][ni][i] + bias[n];
      }
}

// --------------------------------- launcher --------------------------------
extern "C" void kernel_launch(void* const* d_in, const int* in_sizes, int n_in,
                              void* d_out, int out_size, void* d_ws,
                              size_t ws_size, hipStream_t stream) {
  const float* x     = (const float*)d_in[0];
  const float* Wqkv  = (const float*)d_in[1];
  const float* bqkv  = (const float*)d_in[2];
  const float* Wproj = (const float*)d_in[3];
  const float* bproj = (const float*)d_in[4];
  float* out = (float*)d_out;

  // workspace carve-up (f16 buffers), all 256B aligned
  char* ws = (char*)d_ws;
  _Float16* XH     = (_Float16*)(ws + 0);                 // 4.19M elems
  _Float16* WQKVH  = (_Float16*)(ws + (8u << 20));        // 3.15M
  _Float16* WPROJH = (_Float16*)(ws + (14u << 20));       // 1.05M
  _Float16* QH     = (_Float16*)(ws + (16u << 20));       // 4.19M
  _Float16* KH     = (_Float16*)(ws + (25u << 20));
  _Float16* VH     = (_Float16*)(ws + (34u << 20));
  _Float16* OH     = (_Float16*)(ws + (43u << 20));

  const int NX = Mtot * Cc;          // 4194304
  const int NWQKV = Cc * N3C;        // 3145728
  const int NWPROJ = Cc * Cc;        // 1048576

  cvt_f32_f16<<<2048, 256, 0, stream>>>(x, XH, NX);
  cvt_f32_f16<<<2048, 256, 0, stream>>>(Wqkv, WQKVH, NWQKV);
  cvt_f32_f16<<<2048, 256, 0, stream>>>(Wproj, WPROJH, NWPROJ);

  qkv_gemm<<<dim3(Mtot / 128, N3C / 128), 256, 0, stream>>>(XH, WQKVH, bqkv,
                                                            QH, KH, VH);
  attn_fwd<<<dim3(Bc * Hc, Tc / 128), 256, 0, stream>>>(QH, KH, VH, OH);
  proj_gemm<<<dim3(Mtot / 128, Cc / 128), 256, 0, stream>>>(OH, WPROJH, bproj,
                                                            out);
}
